// TimeLSTM_8967891714416
// MI455X (gfx1250) — matrix operand order repaired and checked by reading, not compiled, not run on hardware
//
#include <hip/hip_runtime.h>
#include <hip/hip_bf16.h>

typedef __attribute__((ext_vector_type(16))) __bf16 v16bf;
typedef __attribute__((ext_vector_type(8)))  __bf16 v8bf;
typedef __attribute__((ext_vector_type(8)))  float  v8f;

#define BB 64
#define SS 128
#define FF 64
#define HH 64

// Branchless fast transcendentals: v_exp_f32 / v_rcp_f32 TRANS ops, no OCML branches.
__device__ __forceinline__ float fast_rcp(float x){ return __builtin_amdgcn_rcpf(x); }
__device__ __forceinline__ float fast_sigmoid(float x){
  return fast_rcp(1.0f + __expf(-x));
}
__device__ __forceinline__ float fast_tanh(float x){
  // tanh(x) = 1 - 2/(exp(2x)+1); monotone, exact at 0, saturates correctly.
  return 1.0f - 2.0f * fast_rcp(__expf(2.0f * x) + 1.0f);
}

// Build a 16x32 (A) or 32x16 (B) bf16 WMMA fragment from two 16B-contiguous LDS chunks.
__device__ __forceinline__ v16bf make_frag(const __bf16* p0, const __bf16* p1){
  v8bf lo = *(const v8bf*)p0;
  v8bf hi = *(const v8bf*)p1;
  v16bf r;
#pragma unroll
  for (int e=0;e<8;++e){ r[e]=lo[e]; r[e+8]=hi[e]; }
  return r;
}

// ---------------------------------------------------------------------------
// Kernel 1: recurrent scan. grid.x = F*2 (feature, direction). 256 thr = 8 waves.
// Each step: 5 GEMMs [64,64]x[64,64] via v_wmma_f32_16x16x32_bf16 (20 wmma/wave),
// fp32 gate math, h/c state in LDS, alpha partial reduced via ds_add_f32.
// ---------------------------------------------------------------------------
__global__ __launch_bounds__(256)
void tlstm_scan(const float* __restrict__ X,  const float* __restrict__ TM,
                const float* __restrict__ Ut, const float* __restrict__ Ui,
                const float* __restrict__ Uf, const float* __restrict__ Uc,
                const float* __restrict__ Uo,
                const float* __restrict__ Wd, const float* __restrict__ Wi,
                const float* __restrict__ Wf, const float* __restrict__ Wc,
                const float* __restrict__ Wo,
                const float* __restrict__ Wci,const float* __restrict__ Wcf,
                const float* __restrict__ Wco,
                const float* __restrict__ bt, const float* __restrict__ bi,
                const float* __restrict__ bf_,const float* __restrict__ bc,
                const float* __restrict__ bo,
                const float* __restrict__ Falpha,
                float* __restrict__ part)
{
  const int f    = blockIdx.x >> 1;
  const int dir  = blockIdx.x & 1;
  const int tid  = threadIdx.x;
  const int wave = tid >> 5, lane = tid & 31;
  const int half = lane >> 4, ln = lane & 15;
  const int ni   = wave & 3,  ai = wave >> 2;     // wave -> (n-tile, m-tile pair)
  const int mi0  = 2*ai;
  const int kcol = ni*16 + ln;                    // output column owned by this lane

  const int HP = 72;   // bf16 LDS row pitch (multiple of 8 -> 16B aligned rows)
  const int CP = 72;   // f32  LDS row pitch

  __shared__ __bf16 Wswz[5*4096];                 // swizzled bf16 weights (40 KB)
  __shared__ __bf16 h_bf[64*72];                  // h state, bf16 (for WMMA A)
  __shared__ __bf16 c_bf[64*72];                  // c state, bf16 (for W_decomp GEMM)
  __shared__ float  c_f32[64*72];                 // c state, exact fp32
  __shared__ float  x_lds[64], T_lds[64], sT_lds[64], alpha_acc[64];

  for (int i = tid; i < 64*72; i += 256){
    h_bf[i] = (__bf16)0.0f; c_bf[i] = (__bf16)0.0f; c_f32[i] = 0.0f;
  }

  // Swizzle this feature's 5 weight matrices into per-fragment-contiguous bf16.
  // GEMM: out[b,k] = sum_h A[b,h] * W[f,h,k]  => B[kc,n] = W[f*4096 + kc*64 + n]
  const float* Wg[5] = {Wd, Wi, Wf, Wc, Wo};
  for (int idx = tid; idx < 5*4096; idx += 256){
    int g = idx >> 12; int r = idx & 4095;
    int e = r & 15, l = (r>>4)&31, kb = (r>>9)&1, nn = r>>10;
    int k = kb*32 + (l>>4)*8 + (e&7) + (e>>3)*16; // contraction row
    int n = nn*16 + (l&15);                       // output column
    Wswz[idx] = (__bf16)Wg[g][f*4096 + k*64 + n];
  }
  __syncthreads();

  // Park this wave's B fragments in VGPRs for all 128 steps (5 gates x 2 k-blocks).
  v16bf Bfrag[5][2];
#pragma unroll
  for (int g = 0; g < 5; ++g)
#pragma unroll
    for (int kb = 0; kb < 2; ++kb){
      const __bf16* p = &Wswz[g*4096 + ((ni*2+kb)*32 + lane)*16];
      Bfrag[g][kb] = make_frag(p, p+8);
    }

  // Per-lane (column k) loop-invariant parameters.
  const float ut  = Ut [f*64+kcol], ui  = Ui [f*64+kcol], uf  = Uf [f*64+kcol];
  const float uc  = Uc [f*64+kcol], uo  = Uo [f*64+kcol];
  const float wci = Wci[f*64+kcol], wcf = Wcf[f*64+kcol], wco = Wco[f*64+kcol];
  const float cbt = bt [f*64+kcol], cbi = bi [f*64+kcol], cbf = bf_[f*64+kcol];
  const float cbc = bc [f*64+kcol], cbo = bo [f*64+kcol];
  const float fav = Falpha[f*128 + dir*64 + kcol];

  for (int s = 0; s < SS; ++s){
    if (tid < 64){
      int b  = tid;
      int xi = dir ? (SS-1-s) : s;
      int ti = dir ? (s == 0 ? 0 : SS-s) : s;     // t_bwd = concat(t[:1], t[1:][::-1])
      float xv = X [b*SS*FF + xi*FF + f];
      float tv = TM[b*SS*FF + ti*FF + f];
      float Tv = fast_rcp(__logf(tv + 2.7182818284590452f));
      x_lds[b] = xv; T_lds[b] = Tv; sT_lds[b] = fast_sigmoid(Tv); alpha_acc[b] = 0.0f;
    }
    __syncthreads();

    // A fragments: rows of h (gates) and c (decomp), 2 m-tiles x 2 k-blocks.
    v16bf hA[2][2], cA[2][2];
#pragma unroll
    for (int m2 = 0; m2 < 2; ++m2){
      int m = (mi0+m2)*16 + ln;
#pragma unroll
      for (int kb = 0; kb < 2; ++kb){
        const __bf16* hp = &h_bf[m*HP + kb*32 + half*8];
        const __bf16* cp = &c_bf[m*HP + kb*32 + half*8];
        hA[m2][kb] = make_frag(hp, hp+16);
        cA[m2][kb] = make_frag(cp, cp+16);
      }
    }
    __syncthreads();   // fragments read before anyone overwrites the state

    v8f acc[5][2];
#pragma unroll
    for (int g = 0; g < 5; ++g)
#pragma unroll
      for (int m2 = 0; m2 < 2; ++m2){
        v8f z = (v8f)0.0f;
#pragma unroll
        for (int kb = 0; kb < 2; ++kb){
          v16bf A = (g == 0) ? cA[m2][kb] : hA[m2][kb];
          z = __builtin_amdgcn_wmma_f32_16x16x32_bf16(false, A, false, Bfrag[g][kb],
                                                      (short)0, z, false, false);
        }
        acc[g][m2] = z;
      }

    // Elementwise gate math in fp32; C/D layout: row b = tile*16 + half*8 + r.
#pragma unroll
    for (int m2 = 0; m2 < 2; ++m2){
      int b_base = (mi0+m2)*16 + half*8;
#pragma unroll
      for (int r = 0; r < 8; ++r){
        int b = b_base + r;
        float xv = x_lds[b], Tv = T_lds[b], sT = sT_lds[b];
        float cold = c_f32[b*CP + kcol];
        float CST  = fast_tanh(acc[0][m2][r]);
        float c1   = cold - CST + Tv*CST;
        float tg   = fast_sigmoid(xv*ut + sT + cbt);
        float ig   = fast_sigmoid(xv*ui + acc[1][m2][r] + c1*wci + cbi);
        float fg   = fast_sigmoid(xv*uf + acc[2][m2][r] + c1*wcf + cbf);
        float cg   = fast_tanh   (xv*uc + acc[3][m2][r] + cbc);
        float Ct   = (fg + tg)*c1 + ig*tg*cg;
        float og   = fast_sigmoid(xv*uo + acc[4][m2][r] + tg + Ct*wco + cbo);
        float hn   = og * fast_tanh(Ct);
        c_f32[b*CP + kcol] = Ct;
        c_bf [b*HP + kcol] = (__bf16)Ct;
        h_bf [b*HP + kcol] = (__bf16)hn;
        atomicAdd(&alpha_acc[b], hn * fav);       // ds_add_f32: fold F_alpha here
      }
    }
    __syncthreads();

    if (tid < 64)
      part[((dir*FF + f)*SS + s)*BB + tid] = alpha_acc[tid];
  }
}

// ---------------------------------------------------------------------------
// Kernel 2: head. One block per batch b. tanh + LN1 -> [64,128] bf16,
// WMMA GEMM [64,128]x[128,256] (32 wmma/wave), LN2, Phi, max over F.
// ---------------------------------------------------------------------------
__global__ __launch_bounds__(256)
void tlstm_head(const float* __restrict__ part,
                const float* __restrict__ Fab,
                const float* __restrict__ ln1g, const float* __restrict__ ln1b,
                const float* __restrict__ FbW,  const float* __restrict__ Fbb,
                const float* __restrict__ ln2g, const float* __restrict__ ln2b,
                const float* __restrict__ PhiW, const float* __restrict__ Phib,
                float* __restrict__ out)
{
  const int b    = blockIdx.x;
  const int tid  = threadIdx.x;
  const int wave = tid >> 5, lane = tid & 31;
  const int half = lane >> 4, ln = lane & 15;

  __shared__ __align__(16) char smem[16384 + 65536 + 65536];  // 144 KB
  __bf16* aN  = (__bf16*)(smem);                  // [64][128] bf16 (A matrix)
  __bf16* Bsw = (__bf16*)(smem + 16384);          // [32768] swizzled F_beta_w.T
  float*  fb  = (float*)(smem + 16384 + 65536);   // [64][256]
  __shared__ float red[512];
  __shared__ float mu[128];

  // Phase 1: alphas = tanh(fwd+bwd+bias); LN1 over all F*S = 8192 elems of this b.
  float av[32];
  float ls = 0.0f, lq = 0.0f;
#pragma unroll 4
  for (int j = 0; j < 32; ++j){
    int idx = tid + j*256;
    int ff = idx >> 7, ss2 = idx & 127;
    float v = fast_tanh(part[(ff*SS+ss2)*BB + b] + part[((FF+ff)*SS+ss2)*BB + b] + Fab[ff]);
    av[j] = v; ls += v; lq += v*v;
  }
  red[tid] = ls; red[256+tid] = lq;
  __syncthreads();
  for (int off = 128; off > 0; off >>= 1){
    if (tid < off){ red[tid] += red[tid+off]; red[256+tid] += red[256+tid+off]; }
    __syncthreads();
  }
  float mean = red[0]   * (1.0f/8192.0f);
  float var  = red[256] * (1.0f/8192.0f) - mean*mean;
  float inv  = rsqrtf(var + 1e-5f);
#pragma unroll 4
  for (int j = 0; j < 32; ++j){
    int idx = tid + j*256;
    aN[idx] = (__bf16)((av[j]-mean)*inv * ln1g[idx] + ln1b[idx]);
  }

  // Phase 2: swizzle F_beta_w (B[kc=s][n] = FbW[n*128+s]) to bf16 fragments.
  for (int idx = tid; idx < 32768; idx += 256){
    int e = idx & 15, l = (idx>>4)&31, kb = (idx>>9)&3, nn = idx>>11;
    int k = kb*32 + (l>>4)*8 + (e&7) + (e>>3)*16;
    int n = nn*16 + (l&15);
    Bsw[idx] = (__bf16)FbW[n*SS + k];
  }
  __syncthreads();

  // GEMM fb[64,256] = aN[64,128] x B[128,256]; wave w owns n-tiles {2w, 2w+1}.
  v8f acc[4][2];
#pragma unroll
  for (int mi = 0; mi < 4; ++mi)
#pragma unroll
    for (int nj = 0; nj < 2; ++nj) acc[mi][nj] = (v8f)0.0f;

#pragma unroll
  for (int kb = 0; kb < 4; ++kb){
    v16bf Af[4], Bf[2];
#pragma unroll
    for (int mi = 0; mi < 4; ++mi){
      const __bf16* p = &aN[(mi*16+ln)*SS + kb*32 + half*8];
      Af[mi] = make_frag(p, p+16);
    }
#pragma unroll
    for (int nj = 0; nj < 2; ++nj){
      const __bf16* p = &Bsw[((2*wave+nj)*4 + kb)*512 + lane*16];
      Bf[nj] = make_frag(p, p+8);
    }
#pragma unroll
    for (int mi = 0; mi < 4; ++mi)
#pragma unroll
      for (int nj = 0; nj < 2; ++nj)
        acc[mi][nj] = __builtin_amdgcn_wmma_f32_16x16x32_bf16(false, Af[mi], false, Bf[nj],
                                                              (short)0, acc[mi][nj], false, false);
  }

  // Bias + LN2 stats over all 64*256 = 16384 elems.
  float ls2 = 0.0f, lq2 = 0.0f;
#pragma unroll
  for (int mi = 0; mi < 4; ++mi)
#pragma unroll
    for (int nj = 0; nj < 2; ++nj){
      int n = (2*wave+nj)*16 + ln;
#pragma unroll
      for (int r = 0; r < 8; ++r){
        int fr = mi*16 + half*8 + r;
        float v = acc[mi][nj][r] + Fbb[n];
        fb[fr*256 + n] = v;
        ls2 += v; lq2 += v*v;
      }
    }
  __syncthreads();
  red[tid] = ls2; red[256+tid] = lq2;
  __syncthreads();
  for (int off = 128; off > 0; off >>= 1){
    if (tid < off){ red[tid] += red[tid+off]; red[256+tid] += red[256+tid+off]; }
    __syncthreads();
  }
  float mean2 = red[0]   * (1.0f/16384.0f);
  float var2  = red[256] * (1.0f/16384.0f) - mean2*mean2;
  float inv2  = rsqrtf(var2 + 1e-5f);

  // Phi projection (O=2) + max over F.
  if (tid < 128){
    int ff = tid >> 1, o = tid & 1;
    float a0 = 0.0f;
    for (int n = 0; n < 256; ++n){
      float v = (fb[ff*256+n]-mean2)*inv2 * ln2g[ff*256+n] + ln2b[ff*256+n];
      a0 += v * PhiW[o*256+n];
    }
    mu[tid] = a0 + Phib[o];
  }
  __syncthreads();
  if (tid < 2){
    float mx = -3.4e38f;
    for (int ff = 0; ff < FF; ++ff) mx = fmaxf(mx, mu[ff*2+tid]);
    out[b*2+tid] = mx;
  }
}

extern "C" void kernel_launch(void* const* d_in, const int* in_sizes, int n_in,
                              void* d_out, int out_size, void* d_ws, size_t ws_size,
                              hipStream_t stream)
{
  (void)in_sizes; (void)n_in; (void)out_size; (void)ws_size;
  const float* X   = (const float*)d_in[0];   // historic_features [B,S,F]
  const float* TM  = (const float*)d_in[1];   // timestamp        [B,S,F]
  // d_in[2] last_features, d_in[3] features_freqs: unused by reference
  const float* Ut  = (const float*)d_in[4];
  const float* Ui  = (const float*)d_in[5];
  const float* Uf  = (const float*)d_in[6];
  const float* Uc  = (const float*)d_in[7];
  const float* Uo  = (const float*)d_in[8];
  const float* Wd  = (const float*)d_in[9];
  const float* Wi  = (const float*)d_in[10];
  const float* Wf  = (const float*)d_in[11];
  const float* Wc  = (const float*)d_in[12];
  const float* Wo  = (const float*)d_in[13];
  const float* Wci = (const float*)d_in[14];
  const float* Wcf = (const float*)d_in[15];
  const float* Wco = (const float*)d_in[16];
  const float* bt  = (const float*)d_in[17];
  const float* bi  = (const float*)d_in[18];
  const float* bf_ = (const float*)d_in[19];
  const float* bc  = (const float*)d_in[20];
  const float* bo  = (const float*)d_in[21];
  const float* Fa  = (const float*)d_in[22];
  const float* Fab = (const float*)d_in[23];
  const float* FbW = (const float*)d_in[24];
  const float* Fbb = (const float*)d_in[25];
  const float* PhW = (const float*)d_in[26];
  const float* Phb = (const float*)d_in[27];
  const float* l1g = (const float*)d_in[28];
  const float* l1b = (const float*)d_in[29];
  const float* l2g = (const float*)d_in[30];
  const float* l2b = (const float*)d_in[31];

  float* part = (float*)d_ws;  // [2][F][S][B] fp32 = 4 MB alpha partials

  tlstm_scan<<<dim3(FF*2), dim3(256), 0, stream>>>(
      X, TM, Ut, Ui, Uf, Uc, Uo, Wd, Wi, Wf, Wc, Wo,
      Wci, Wcf, Wco, bt, bi, bf_, bc, bo, Fa, part);

  tlstm_head<<<dim3(BB), dim3(256), 0, stream>>>(
      part, Fab, l1g, l1b, FbW, Fbb, l2g, l2b, PhW, Phb, (float*)d_out);
}
